// GTGModule_21818433864405
// MI455X (gfx1250) — matrix operand order repaired and checked by reading, not compiled, not run on hardware
//
#include <hip/hip_runtime.h>
#include <hip/hip_bf16.h>
#include <math.h>

#define NROW 8192
#define DIM  512
#define NC   10
#define NCP  16
#define ITERS 30
#define NLAB 1024
#define EPSF 1e-8f

typedef __attribute__((ext_vector_type(16))) __bf16 v16bf;
typedef __attribute__((ext_vector_type(8)))  __bf16 v8bf;
typedef __attribute__((ext_vector_type(8)))  float  v8f;

static __device__ __forceinline__ v16bf combine16(v8bf lo, v8bf hi) {
    v16bf r;
#pragma unroll
    for (int i = 0; i < 8; ++i) { r[i] = lo[i]; r[i + 8] = hi[i]; }
    return r;
}

// CDNA5 async copy: 16B global -> LDS, tracked by ASYNCcnt.
// VDST = 32-bit LDS byte address (low 32 bits of generic pointer, per the
// LDS aperture rule), VADDR = 64-bit global address.
static __device__ __forceinline__ void async_cp16(const __bf16* g, __bf16* l) {
    const unsigned lds = (unsigned)(unsigned long long)l;
    asm volatile("global_load_async_to_lds_b128 %0, %1, off"
                 :: "v"(lds), "v"((const void*)g)
                 : "memory");
}
static __device__ __forceinline__ void async_wait0() {
    asm volatile("s_wait_asynccnt 0" ::: "memory");
}

// ---------------------------------------------------------------------------
// 1) Row-normalize embedding (fp32 in) -> bf16 En.  One wave per row.
// ---------------------------------------------------------------------------
__global__ void k_normalize(const float* __restrict__ E, __bf16* __restrict__ EnB) {
    const int row  = blockIdx.x;
    const int lane = threadIdx.x;                 // 0..31
    const float* e = E + (size_t)row * DIM + lane * 16;
    float v[16];
#pragma unroll
    for (int i = 0; i < 4; ++i) {
        float4 f = *(const float4*)(e + i * 4);
        v[i*4+0] = f.x; v[i*4+1] = f.y; v[i*4+2] = f.z; v[i*4+3] = f.w;
    }
    float s = 0.f;
#pragma unroll
    for (int i = 0; i < 16; ++i) s += v[i] * v[i];
#pragma unroll
    for (int m = 16; m >= 1; m >>= 1) s += __shfl_xor(s, m, 32);
    const float scale = 1.0f / (sqrtf(s) + EPSF);

    __bf16* o = EnB + (size_t)row * DIM + lane * 16;
    v8bf lo, hi;
#pragma unroll
    for (int i = 0; i < 8; ++i) { lo[i] = (__bf16)(v[i] * scale); hi[i] = (__bf16)(v[i+8] * scale); }
    *(v8bf*)(o)     = lo;
    *(v8bf*)(o + 8) = hi;
}

// ---------------------------------------------------------------------------
// 2) A = clamp(En En^T,0,1), diag=1, stored bf16; per-block partial sums for
//    the mean.  Block = 8 waves; wave w computes rows [by*128+16w, +16) x
//    cols [bx*128, +128) as 8 WMMA accum tiles.
// ---------------------------------------------------------------------------
__global__ void __launch_bounds__(256) k_abuild(const __bf16* __restrict__ EnB,
                                                __bf16* __restrict__ A,
                                                float* __restrict__ partial) {
    const int lane = threadIdx.x & 31;
    const int wave = threadIdx.x >> 5;            // 0..7
    const int lm = lane & 15, lh = lane >> 4;
    const int r0 = blockIdx.y * 128 + wave * 16;
    const int c0 = blockIdx.x * 128;

    v8f acc[8] = {};
    const __bf16* arow = EnB + (size_t)(r0 + lm) * DIM;

    for (int kt = 0; kt < 16; ++kt) {
        const int k = kt * 32;
        v16bf af = combine16(*(const v8bf*)(arow + k + lh * 8),
                             *(const v8bf*)(arow + k + 16 + lh * 8));
#pragma unroll
        for (int j = 0; j < 8; ++j) {
            const __bf16* brow = EnB + (size_t)(c0 + 16 * j + lm) * DIM + k + lh * 8;
            v16bf bfr = combine16(*(const v8bf*)brow, *(const v8bf*)(brow + 16));
            acc[j] = __builtin_amdgcn_wmma_f32_16x16x32_bf16(
                false, af, false, bfr, (short)0, acc[j], false, false);
        }
    }

    // clamp / diag / store bf16 / accumulate partial mean
    float lsum = 0.f;
    const int mb = lh * 8;
#pragma unroll
    for (int j = 0; j < 8; ++j) {
        const int cb = c0 + 16 * j + lm;          // this lane's column
#pragma unroll
        for (int vv = 0; vv < 8; ++vv) {
            const int r = r0 + mb + vv;
            float x = acc[j][vv];
            x = fminf(fmaxf(x, 0.0f), 1.0f);
            if (r == cb) x = 1.0f;
            lsum += x;
            A[(size_t)r * NROW + cb] = (__bf16)x;
        }
    }
#pragma unroll
    for (int m = 16; m >= 1; m >>= 1) lsum += __shfl_xor(lsum, m, 32);
    __shared__ float wsum[8];
    if (lane == 0) wsum[wave] = lsum;
    __syncthreads();
    if (threadIdx.x == 0) {
        float s = 0.f;
#pragma unroll
        for (int i = 0; i < 8; ++i) s += wsum[i];
        partial[blockIdx.y * gridDim.x + blockIdx.x] = s;
    }
}

// ---------------------------------------------------------------------------
// 3) Deterministic final reduction: 4096 partials -> threshold (mean of A)
// ---------------------------------------------------------------------------
__global__ void k_reduce(const float* __restrict__ partial, float* __restrict__ thr) {
    __shared__ float sm[256];
    const int t = threadIdx.x;
    float s = 0.f;
#pragma unroll
    for (int i = 0; i < 16; ++i) s += partial[t * 16 + i];
    sm[t] = s;
    __syncthreads();
    for (int off = 128; off >= 1; off >>= 1) {
        if (t < off) sm[t] += sm[t + off];
        __syncthreads();
    }
    if (t == 0) thr[0] = sm[0] / ((float)NROW * (float)NROW);
}

// ---------------------------------------------------------------------------
// 4) Zero entries of A below the mean (bf16, 8 elems/thread, b128 ld/st)
// ---------------------------------------------------------------------------
__global__ void k_thresh(__bf16* __restrict__ A, const float* __restrict__ thr) {
    const float th = thr[0];
    const size_t idx = ((size_t)blockIdx.x * blockDim.x + threadIdx.x) * 8;
    v8bf v = *(const v8bf*)(A + idx);
#pragma unroll
    for (int i = 0; i < 8; ++i)
        if ((float)v[i] < th) v[i] = (__bf16)0.0f;
    *(v8bf*)(A + idx) = v;
}

// ---------------------------------------------------------------------------
// 5) Init X0: labeled one-hot, unlabeled uniform.  Also builds Xt (16 x N bf16)
// ---------------------------------------------------------------------------
__global__ void k_initx(const int* __restrict__ lab,
                        float* __restrict__ Xf, __bf16* __restrict__ Xt) {
    const int i = blockIdx.x * blockDim.x + threadIdx.x;
    if (i >= NROW) return;
    float p[NC];
    if (i < NLAB) {
        const int l = lab[i];
#pragma unroll
        for (int c = 0; c < NC; ++c) p[c] = (c == l) ? 1.0f : 0.0f;
    } else {
#pragma unroll
        for (int c = 0; c < NC; ++c) p[c] = 0.1f;
    }
#pragma unroll
    for (int c = 0; c < NC; ++c) Xf[(size_t)i * NC + c] = p[c];
#pragma unroll
    for (int c = 0; c < NCP; ++c)
        Xt[(size_t)c * NROW + i] = (__bf16)((c < NC) ? p[c] : 0.0f);
}

// ---------------------------------------------------------------------------
// 6) One replicator iteration.  Block = 8 waves / 64 rows.
//    X^T K-chunks (16 classes x 512 K) are async-staged into LDS once per
//    block (double-buffered, global_load_async_to_lds_b128) and shared by all
//    waves: 4 row-strips x 2 K-halves, 2 parity accumulators per wave.
// ---------------------------------------------------------------------------
#define CHUNK_K 512
#define XROW    (CHUNK_K + 8)   // +8 bf16 pad -> LDS bank stagger

__global__ void __launch_bounds__(256) k_iter(const __bf16* __restrict__ A,
                                              const __bf16* __restrict__ Xt_in,
                                              const float* __restrict__ Xf_in,
                                              __bf16* __restrict__ Xt_out,
                                              float* __restrict__ Xf_out,
                                              float* __restrict__ xhist,
                                              float* __restrict__ enthist,
                                              int iter) {
    const int lane = threadIdx.x & 31;
    const int wave = threadIdx.x >> 5;            // 0..7
    const int lm = lane & 15, lh = lane >> 4;
    const int wrow = wave & 3;                    // row strip within block
    const int wk   = wave >> 2;                   // K half: 0 or 1
    const int r0 = blockIdx.x * 64;
    const int rows0 = r0 + wrow * 16;

    __shared__ __align__(16) __bf16 xt[2][16][XROW];
    __shared__ float yt[2][64][16];

    v8f acc0 = {}, acc1 = {};
    const __bf16* arow = A + (size_t)(rows0 + lm) * NROW;

    // stage K-chunk kc of X^T into LDS buffer b: 16 rows x 512 bf16 = 16 KB,
    // 256 threads x 4 async b128 copies each.
    auto stage = [&](int kc, int b) {
        const int t = threadIdx.x;
#pragma unroll
        for (int q = 0; q < 4; ++q) {
            const int u   = q * 256 + t;          // 16B unit id, 0..1023
            const int c   = u >> 6;               // class row 0..15
            const int off = (u & 63) * 8;         // element offset 0..504
            async_cp16(Xt_in + (size_t)c * NROW + kc * CHUNK_K + off,
                       &xt[b][c][off]);
        }
    };

    stage(0, 0);
    int buf = 0;
    for (int kc = 0; kc < NROW / CHUNK_K; ++kc) {
        async_wait0();
        __syncthreads();                          // staged chunk visible to all
        if (kc + 1 < NROW / CHUNK_K) stage(kc + 1, buf ^ 1);
        const int kbase = kc * CHUNK_K;
#pragma unroll
        for (int j = 0; j < 8; ++j) {
            const int kl = (wk * 8 + j) * 32;     // local K offset in chunk
            const int kg = kbase + kl;            // global K
            if (j == 0) __builtin_prefetch(arow + kg + CHUNK_K, 0, 1);
            v16bf af = combine16(*(const v8bf*)(arow + kg + lh * 8),
                                 *(const v8bf*)(arow + kg + 16 + lh * 8));
            const __bf16* bp = &xt[buf][lm][kl + lh * 8];
            v16bf bfr = combine16(*(const v8bf*)bp, *(const v8bf*)(bp + 16));
            if (j & 1)
                acc1 = __builtin_amdgcn_wmma_f32_16x16x32_bf16(
                    false, af, false, bfr, (short)0, acc1, false, false);
            else
                acc0 = __builtin_amdgcn_wmma_f32_16x16x32_bf16(
                    false, af, false, bfr, (short)0, acc0, false, false);
        }
        buf ^= 1;
    }

    // combine parity accumulators; deposit this wave's 16x16 partial tile
#pragma unroll
    for (int vv = 0; vv < 8; ++vv)
        yt[wk][wrow * 16 + 8 * lh + vv][lm] = acc0[vv] + acc1[vv];
    __syncthreads();

    const int t = threadIdx.x;
    if (t < 64) {
        const int row = r0 + t;
        float xn[NC];
        float s = 0.f;
#pragma unroll
        for (int c = 0; c < NC; ++c) {
            const float y = yt[0][t][c] + yt[1][t][c];   // fixed-order K merge
            const float x = Xf_in[(size_t)row * NC + c] * y;
            xn[c] = x; s += x;
        }
        const float inv = 1.0f / (s + EPSF);
        float ent = 0.f;
#pragma unroll
        for (int c = 0; c < NC; ++c) {
            const float x = xn[c] * inv;
            xn[c] = x;
            ent -= x * logf(x + EPSF);
            xhist[(size_t)row * (ITERS * NC) + iter * NC + c] = x;
            Xf_out[(size_t)row * NC + c] = x;
        }
        enthist[(size_t)row * ITERS + iter] = ent * 0.43429448190325176f; // 1/ln(10)
#pragma unroll
        for (int c = 0; c < NCP; ++c)
            Xt_out[(size_t)c * NROW + row] = (__bf16)((c < NC) ? xn[c] : 0.0f);
    }
}

// ---------------------------------------------------------------------------
extern "C" void kernel_launch(void* const* d_in, const int* in_sizes, int n_in,
                              void* d_out, int out_size, void* d_ws, size_t ws_size,
                              hipStream_t stream) {
    const float* emb = (const float*)d_in[0];
    const int*   lab = (const int*)d_in[1];

    char* ws = (char*)d_ws;
    size_t off = 0;
    __bf16* EnB = (__bf16*)(ws + off); off += (size_t)NROW * DIM * 2;    // 8 MB
    __bf16* A   = (__bf16*)(ws + off); off += (size_t)NROW * NROW * 2;   // 128 MB (L2-resident)
    float* partial = (float*)(ws + off); off += 4096 * 4;
    float* thr     = (float*)(ws + off); off += 256;
    float* Xf0 = (float*)(ws + off); off += (size_t)NROW * NC * 4;
    float* Xf1 = (float*)(ws + off); off += (size_t)NROW * NC * 4;
    __bf16* Xt0 = (__bf16*)(ws + off); off += (size_t)NCP * NROW * 2;
    __bf16* Xt1 = (__bf16*)(ws + off); off += (size_t)NCP * NROW * 2;

    float* xhist   = (float*)d_out;
    float* enthist = xhist + (size_t)NROW * ITERS * NC;

    k_normalize<<<NROW, 32, 0, stream>>>(emb, EnB);

    dim3 gA(64, 64);
    k_abuild<<<gA, 256, 0, stream>>>(EnB, A, partial);
    k_reduce<<<1, 256, 0, stream>>>(partial, thr);
    k_thresh<<<32768, 256, 0, stream>>>(A, thr);

    k_initx<<<NROW / 256, 256, 0, stream>>>(lab, Xf0, Xt0);

    for (int it = 0; it < ITERS; ++it) {
        const __bf16* Xti = (it & 1) ? Xt1 : Xt0;
        __bf16*       Xto = (it & 1) ? Xt0 : Xt1;
        const float*  Xfi = (it & 1) ? Xf1 : Xf0;
        float*        Xfo = (it & 1) ? Xf0 : Xf1;
        k_iter<<<NROW / 64, 256, 0, stream>>>(A, Xti, Xfi, Xto, Xfo,
                                              xhist, enthist, it);
    }
}
